// SelfAttention2d_45561013076265
// MI455X (gfx1250) — matrix operand order repaired and checked by reading, not compiled, not run on hardware
//
#include <hip/hip_runtime.h>
#include <math.h>

typedef _Float16 f16;
typedef __attribute__((ext_vector_type(16))) _Float16 v16h;
typedef __attribute__((ext_vector_type(8)))  _Float16 v8h;
typedef __attribute__((ext_vector_type(8)))  float    v8f;

// Problem constants (fixed by the reference):
// b=2, c=256, h=w=64 -> n=4096 tokens, 8 heads, head_dim=32, 8 groups.
#define NTOK   4096
#define CCH    256
#define HEADS  8
#define HD     32
#define BATCH  2

__device__ __forceinline__ v16h join16(v8h lo, v8h hi8) {
  v16h r;
#pragma unroll
  for (int i = 0; i < 8; ++i) { r[i] = lo[i]; r[i + 8] = hi8[i]; }
  return r;
}

// A-operand (16xK, K=32 slice) from row-major [M][K] f16:
// lane l holds row M=l; K = hi*8+[0,8) and 16+hi*8+[0,8).
__device__ __forceinline__ v16h load_a(const f16* base, int ld, int l, int hi) {
  const v8h* p0 = (const v8h*)(base + (size_t)l * ld + hi * 8);
  const v8h* p1 = (const v8h*)(base + (size_t)l * ld + 16 + hi * 8);
  return join16(*p0, *p1);
}

// B-operand (32xN, N=16) from [N][K] (B-transposed) row-major f16:
// lane n holds col N=n; K = hi*8+[0,8) and 16+hi*8+[0,8).
__device__ __forceinline__ v16h load_b(const f16* base, int ld, int n, int hi) {
  const v8h* p0 = (const v8h*)(base + (size_t)n * ld + hi * 8);
  const v8h* p1 = (const v8h*)(base + (size_t)n * ld + 16 + hi * 8);
  return join16(*p0, *p1);
}

// CDNA5 async global->LDS copy (16B per lane), tracked by ASYNCcnt.
// VDST = 32-bit LDS byte offset (generic LDS addresses carry it in addr[31:0]).
__device__ __forceinline__ void async_b128(void* lds, const void* g) {
  unsigned lo = (unsigned)(uintptr_t)lds;
  asm volatile("global_load_async_to_lds_b128 %0, %1, off"
               :: "v"(lo), "v"(g) : "memory");
}
__device__ __forceinline__ void wait_async0() {
  asm volatile("s_wait_asynccnt 0x0" ::: "memory");
}

// ---------------- GroupNorm ----------------
__global__ __launch_bounds__(256) void gn_stats(const float* __restrict__ x,
                                                float* __restrict__ stats) {
  const int bg = blockIdx.x;                       // 0..15
  const float* p = x + (size_t)bg * (32 * NTOK);
  float s = 0.f, s2 = 0.f;
  for (int i = threadIdx.x; i < 32 * NTOK; i += 256) {
    float v = p[i]; s += v; s2 += v * v;
  }
  __shared__ float rs[256], rq[256];
  rs[threadIdx.x] = s; rq[threadIdx.x] = s2;
  __syncthreads();
  for (int o = 128; o > 0; o >>= 1) {
    if (threadIdx.x < o) { rs[threadIdx.x] += rs[threadIdx.x + o];
                           rq[threadIdx.x] += rq[threadIdx.x + o]; }
    __syncthreads();
  }
  if (threadIdx.x == 0) {
    const float inv = 1.f / (32.f * NTOK);
    float m = rs[0] * inv;
    float var = rq[0] * inv - m * m;
    stats[bg * 2 + 0] = m;
    stats[bg * 2 + 1] = rsqrtf(var + 1e-5f);
  }
}

// Normalize + affine + transpose to token-major fp16: xh[(b*4096+t)*256 + c]
__global__ __launch_bounds__(256) void gn_norm(const float* __restrict__ x,
                                               const float* __restrict__ gamma,
                                               const float* __restrict__ beta,
                                               const float* __restrict__ stats,
                                               f16* __restrict__ xh) {
  int idx = blockIdx.x * 256 + threadIdx.x;        // over 2*4096*256 = 2^21
  int c = idx & 255;
  int t = (idx >> 8) & 4095;
  int b = idx >> 20;
  int g = c >> 5;
  float m = stats[(b * 8 + g) * 2 + 0];
  float r = stats[(b * 8 + g) * 2 + 1];
  float v = (x[((size_t)(b * CCH + c)) * NTOK + t] - m) * r * gamma[c] + beta[c];
  xh[idx] = (f16)v;
}

// Convert weights to fp16 (layouts already [N][K] = B-operand friendly).
__global__ __launch_bounds__(256) void wconv(const float* __restrict__ qkvw,
                                             const float* __restrict__ projw,
                                             f16* __restrict__ qkvwh,
                                             f16* __restrict__ projwh) {
  int i = blockIdx.x * 256 + threadIdx.x;
  if (i < 3 * CCH * CCH) qkvwh[i] = (f16)qkvw[i];
  if (i < CCH * CCH)     projwh[i] = (f16)projw[i];
}

// ---------------- QKV GEMM ----------------
// Y[8192 x 768] = xh[8192 x 256] * qkv_w^T.
// Q,K scattered to [b][head][token][32]; V scattered TRANSPOSED to
// [b][head][32][token] so the attention kernel can stage V^T tiles with
// plain contiguous 128-bit async copies.
__global__ __launch_bounds__(128) void qkv_gemm(const f16* __restrict__ xh,
                                                const f16* __restrict__ wh,
                                                const float* __restrict__ bias,
                                                f16* __restrict__ qh,
                                                f16* __restrict__ kh,
                                                f16* __restrict__ vtg) {
  const int wav = threadIdx.x >> 5, lane = threadIdx.x & 31;
  const int l = lane & 15, hi = lane >> 4;
  const int tile = blockIdx.x * 4 + wav;           // 512 * 48 tiles
  const int mt = tile & 511, nt = tile >> 9;
  const f16* A = xh + (size_t)(mt * 16) * CCH;
  const f16* B = wh + (size_t)(nt * 16) * CCH;
  v8f acc = {};
#pragma unroll
  for (int kk = 0; kk < CCH; kk += 32) {
    v16h a = load_a(A + kk, CCH, l, hi);
    v16h b = load_b(B + kk, CCH, l, hi);
    acc = __builtin_amdgcn_wmma_f32_16x16x32_f16(false, a, false, b,
                                                 (short)0, acc, false, false);
  }
  const int o = nt * 16 + l;                       // output channel 0..767
  const float bo = bias[o];
  const int sec = o >> 8;                          // 0=q 1=k 2=v
  const int head = (o >> 5) & 7;
  const int hd = o & 31;
#pragma unroll
  for (int r = 0; r < 8; ++r) {
    int gm = mt * 16 + r + 8 * hi;                 // global token row
    int b_ = gm >> 12, t = gm & 4095;
    f16 v = (f16)(acc[r] + bo);
    size_t bh = (size_t)(b_ * HEADS + head);
    if (sec == 0)      qh[(bh * NTOK + t) * HD + hd] = v;
    else if (sec == 1) kh[(bh * NTOK + t) * HD + hd] = v;
    else               vtg[(bh * HD + hd) * NTOK + t] = v;   // transposed
  }
}

// ---------------- Flash attention ----------------
// Block: one (b,head) x 64-query slab; 4 waves, each owns 16 query rows.
// K/V^T tiles double-buffered in LDS via async global->LDS copies.
#define KB   64
#define KSTR 40  // padded [token][hd] row stride (80B: bank-spread, 16B align)
#define VSTR 72  // padded [hd][token] row stride (144B)
#define PSTR 72  // padded P scratch row stride

__global__ __launch_bounds__(128) void attn(const f16* __restrict__ qh,
                                            const f16* __restrict__ kh,
                                            const f16* __restrict__ vtg,
                                            f16* __restrict__ aout) {
  __shared__ __align__(16) f16 kT[2][KB * KSTR];   // [token][hd]
  __shared__ __align__(16) f16 vT[2][32 * VSTR];   // [hd][token]
  __shared__ __align__(16) f16 pT[4 * 16 * PSTR];  // per-wave P scratch [M][K]

  const int bh = blockIdx.x >> 6;                  // 0..15
  const int qblk = blockIdx.x & 63;
  const int wav = threadIdx.x >> 5, lane = threadIdx.x & 31;
  const int l = lane & 15, hi = lane >> 4;

  const f16* Q  = qh  + ((size_t)bh * NTOK + qblk * 64 + wav * 16) * HD;
  const f16* K  = kh  + (size_t)bh * NTOK * HD;      // [token][32]
  const f16* VT = vtg + (size_t)bh * HD * NTOK;      // [32][token]

  const v16h qa = load_a(Q, HD, l, hi);            // Q tile lives in registers
  v8f o0 = {}, o1 = {};
  float m[8], sl[8];
#pragma unroll
  for (int r = 0; r < 8; ++r) { m[r] = -__builtin_inff(); sl[r] = 0.f; }
  const float scale = 0.17677669529663687f;        // 1/sqrt(32)
  f16* pW = pT + wav * 16 * PSTR;

  // Issue async stage of one K/V^T block into LDS buffer `buf`.
  auto stage = [&](int buf, int kb) {
    // K tile: 64 rows x 32 halves -> 4 chunks/row, 256 chunks, 2 per thread.
    for (int ch = threadIdx.x; ch < 256; ch += 128) {
      int tok = ch >> 2, c = ch & 3;
      async_b128(&kT[buf][tok * KSTR + c * 8], K + ((size_t)kb + tok) * HD + c * 8);
    }
    // V^T tile: 32 rows x 64 halves -> 8 chunks/row, 256 chunks, 2 per thread.
    for (int ch = threadIdx.x; ch < 256; ch += 128) {
      int d = ch >> 3, c = ch & 7;
      async_b128(&vT[buf][d * VSTR + c * 8], VT + (size_t)d * NTOK + kb + c * 8);
    }
  };
  stage(0, 0);

  for (int it = 0; it < NTOK / KB; ++it) {
    const int cur = it & 1;
    wait_async0();                 // this wave's copies into `cur` landed
    __syncthreads();               // everyone's landed; prev buffer released
    if (it + 1 < NTOK / KB) stage(1 - cur, (it + 1) * KB);

    // S = Q * K^T : four 16x16 tiles, K-dim = hd = 32 in one WMMA each.
    v8f s[4];
#pragma unroll
    for (int t = 0; t < 4; ++t) {
      v16h bk = load_b(&kT[cur][t * 16 * KSTR], KSTR, l, hi);
      v8f z = {};
      s[t] = __builtin_amdgcn_wmma_f32_16x16x32_f16(false, qa, false, bk,
                                                    (short)0, z, false, false);
    }
    // Online softmax (rows live across 16 lanes of each half-wave).
#pragma unroll
    for (int r = 0; r < 8; ++r) {
      float rm = -__builtin_inff();
#pragma unroll
      for (int t = 0; t < 4; ++t) {
        float sv = s[t][r] * scale; s[t][r] = sv; rm = fmaxf(rm, sv);
      }
#pragma unroll
      for (int off = 1; off < 16; off <<= 1)
        rm = fmaxf(rm, __shfl_xor(rm, off, 16));
      float mn = fmaxf(m[r], rm);
      float al = __expf(m[r] - mn);
      m[r] = mn;
      float rs = 0.f;
#pragma unroll
      for (int t = 0; t < 4; ++t) {
        float p = __expf(s[t][r] - mn);
        rs += p;
        pW[(r + 8 * hi) * PSTR + t * 16 + l] = (f16)p;  // C-layout -> [M][K]
      }
#pragma unroll
      for (int off = 1; off < 16; off <<= 1)
        rs += __shfl_xor(rs, off, 16);
      sl[r] = sl[r] * al + rs;
      o0[r] *= al; o1[r] *= al;
    }
    // Per-wave LDS round-trip to relayout P into A-operand form.
    asm volatile("s_wait_dscnt 0x0" ::: "memory");
#pragma unroll
    for (int kc = 0; kc < 2; ++kc) {
      v16h pa = load_a(pW + kc * 32, PSTR, l, hi);
      v16h b0 = load_b(&vT[cur][kc * 32], VSTR, l, hi);
      v16h b1 = load_b(&vT[cur][16 * VSTR + kc * 32], VSTR, l, hi);
      o0 = __builtin_amdgcn_wmma_f32_16x16x32_f16(false, pa, false, b0,
                                                  (short)0, o0, false, false);
      o1 = __builtin_amdgcn_wmma_f32_16x16x32_f16(false, pa, false, b1,
                                                  (short)0, o1, false, false);
    }
  }
  // Epilogue: normalize by row sums, write token-major fp16 for proj GEMM.
  const int b = bh >> 3, h = bh & 7;
  const int q0 = qblk * 64 + wav * 16;
#pragma unroll
  for (int r = 0; r < 8; ++r) {
    float inv = 1.f / sl[r];
    int tok = q0 + r + 8 * hi;
    size_t rowb = ((size_t)(b * NTOK + tok)) * CCH + h * HD;
    aout[rowb + l]      = (f16)(o0[r] * inv);
    aout[rowb + 16 + l] = (f16)(o1[r] * inv);
  }
}

// ---------------- Projection + bias + residual ----------------
__global__ __launch_bounds__(128) void proj_gemm(const f16* __restrict__ ah,
                                                 const f16* __restrict__ wh,
                                                 const float* __restrict__ bias,
                                                 const float* __restrict__ x,
                                                 float* __restrict__ out) {
  const int wav = threadIdx.x >> 5, lane = threadIdx.x & 31;
  const int l = lane & 15, hi = lane >> 4;
  const int tile = blockIdx.x * 4 + wav;           // 512 * 16 tiles
  const int mt = tile & 511, nt = tile >> 9;
  const f16* A = ah + (size_t)(mt * 16) * CCH;
  const f16* B = wh + (size_t)(nt * 16) * CCH;
  v8f acc = {};
#pragma unroll
  for (int kk = 0; kk < CCH; kk += 32) {
    v16h a = load_a(A + kk, CCH, l, hi);
    v16h b = load_b(B + kk, CCH, l, hi);
    acc = __builtin_amdgcn_wmma_f32_16x16x32_f16(false, a, false, b,
                                                 (short)0, acc, false, false);
  }
  const int o = nt * 16 + l;
  const float bo = bias[o];
#pragma unroll
  for (int r = 0; r < 8; ++r) {
    int gm = mt * 16 + r + 8 * hi;
    int b_ = gm >> 12, t = gm & 4095;
    size_t oi = ((size_t)(b_ * CCH + o)) * NTOK + t;
    out[oi] = acc[r] + bo + x[oi];
  }
}

extern "C" void kernel_launch(void* const* d_in, const int* in_sizes, int n_in,
                              void* d_out, int out_size, void* d_ws, size_t ws_size,
                              hipStream_t stream) {
  const float* x     = (const float*)d_in[0];
  const float* gamma = (const float*)d_in[1];
  const float* beta  = (const float*)d_in[2];
  const float* qkvw  = (const float*)d_in[3];
  const float* qkvb  = (const float*)d_in[4];
  const float* projw = (const float*)d_in[5];
  const float* projb = (const float*)d_in[6];
  float* out = (float*)d_out;

  char* w = (char*)d_ws;
  size_t off = 0;
  auto carve = [&](size_t bytes) {
    void* p = w + off;
    off += (bytes + 255) & ~(size_t)255;
    return p;
  };
  float* stats  = (float*)carve(16 * 2 * sizeof(float));
  f16* xh       = (f16*)carve((size_t)BATCH * NTOK * CCH * 2);      // 4 MB
  f16* qkvwh    = (f16*)carve((size_t)3 * CCH * CCH * 2);
  f16* projwh   = (f16*)carve((size_t)CCH * CCH * 2);
  f16* qh       = (f16*)carve((size_t)BATCH * HEADS * NTOK * HD * 2);
  f16* kh       = (f16*)carve((size_t)BATCH * HEADS * NTOK * HD * 2);
  f16* vtg      = (f16*)carve((size_t)BATCH * HEADS * HD * NTOK * 2);
  f16* aout     = (f16*)carve((size_t)BATCH * NTOK * CCH * 2);

  wconv<<<(3 * CCH * CCH + 255) / 256, 256, 0, stream>>>(qkvw, projw, qkvwh, projwh);
  gn_stats<<<BATCH * 8, 256, 0, stream>>>(x, stats);
  gn_norm<<<(BATCH * NTOK * CCH) / 256, 256, 0, stream>>>(x, gamma, beta, stats, xh);
  qkv_gemm<<<(512 * 48) / 4, 128, 0, stream>>>(xh, qkvwh, qkvb, qh, kh, vtg);
  attn<<<16 * 64, 128, 0, stream>>>(qh, kh, vtg, aout);
  proj_gemm<<<(512 * 16) / 4, 128, 0, stream>>>(aout, projwh, projb, x, out);
}
